// BaseAttention_8718783611588
// MI455X (gfx1250) — compile-verified, matrix operand order checked by reading, not run on hardware
//
#include <hip/hip_runtime.h>

// ---------------------------------------------------------------------------
// Sliding-window attention for MI455X (gfx1250, wave32, WMMA).
//   B=2, T=2048, C=1024, H=16, D=64, WINDOW=+-128
// QKV GEMM (25.8 GFLOP) dominates; all matmuls use v_wmma_f32_16x16x32_bf16
// with f32 accumulation. V is stored transposed [B,H,D,T] by the GEMM
// epilogue so every WMMA fragment in the attention kernel is two b128 loads.
// GEMM wave tile = 32x64 (2x4 accumulators): 12 b128 loads per 8 WMMAs.
// ---------------------------------------------------------------------------

#define HEADS 16
#define DH    64
#define TLEN  2048
#define BSZ   2
#define CDIM  1024
#define WIN   128

typedef __attribute__((ext_vector_type(16))) __bf16 bf16x16;
typedef __attribute__((ext_vector_type(8)))  __bf16 bf16x8;
typedef __attribute__((ext_vector_type(4)))  __bf16 bf16x4;
typedef __attribute__((ext_vector_type(8)))  float  f32x8;
typedef __attribute__((ext_vector_type(4)))  float  f32x4;

#define WMMA_BF16(A, Bm, Cc) \
  __builtin_amdgcn_wmma_f32_16x16x32_bf16(false, (A), false, (Bm), (short)0, (Cc), false, false)

// XOR-swap within 16-lane halves (group-of-32 ds_swizzle: offset = xor<<10 | 0x1F)
#define SWZ(x, imm) __int_as_float(__builtin_amdgcn_ds_swizzle(__float_as_int(x), (imm)))

// 16-bit A/B fragment: lane holds two contiguous 8-element K-chunks at +0 and +16.
static __device__ __forceinline__ bf16x16 load_frag(const __bf16* p) {
  bf16x8 lo = *(const bf16x8*)(p);
  bf16x8 hi = *(const bf16x8*)(p + 16);
  return __builtin_shufflevector(lo, hi, 0,1,2,3,4,5,6,7,8,9,10,11,12,13,14,15);
}

// ---------------- fp32 -> bf16 conversion ----------------
__global__ void cvt4_kernel(const float* __restrict__ src, __bf16* __restrict__ dst, int n) {
  int i = (blockIdx.x * blockDim.x + threadIdx.x) * 4;
  if (i < n) {
    f32x4 f = *(const f32x4*)(src + i);
    *(bf16x4*)(dst + i) = __builtin_convertvector(f, bf16x4);
  }
}

// ---------------- QKV projection GEMM ----------------
// out[m,n] = sum_k x[m,k] * Wqkv[n,k];  M=4096, N=3072, K=1024.
// Block = 8 waves laid out 4(M) x 2(N); wave computes 32x64 via 2x4 WMMA
// accumulators -> workgroup tile 128x128.
__global__ __launch_bounds__(256) void qkv_gemm_kernel(
    const __bf16* __restrict__ xb, const __bf16* __restrict__ wb,
    __bf16* __restrict__ qbuf, __bf16* __restrict__ kbuf, __bf16* __restrict__ vtbuf) {
  const int lane  = threadIdx.x & 31;
  const int wave  = threadIdx.x >> 5;
  const int lm    = lane & 15;
  const int hi8   = (lane >> 4) * 8;   // row offset for C layout, K-chunk base for A/B
  const int m0 = blockIdx.y * 128 + (wave & 3) * 32;
  const int n0 = blockIdx.x * 128 + (wave >> 2) * 64;

  f32x8 acc[2][4] = {};

  const __bf16* a0p = xb + (size_t)(m0 + lm) * CDIM + hi8;
  const __bf16* a1p = a0p + (size_t)16 * CDIM;
  const __bf16* bp0 = wb + (size_t)(n0 + lm) * CDIM + hi8;
  const __bf16* bp1 = bp0 + (size_t)16 * CDIM;
  const __bf16* bp2 = bp0 + (size_t)32 * CDIM;
  const __bf16* bp3 = bp0 + (size_t)48 * CDIM;

  for (int k0 = 0; k0 < CDIM; k0 += 32) {
    bf16x16 A0 = load_frag(a0p + k0);
    bf16x16 A1 = load_frag(a1p + k0);
    bf16x16 B0 = load_frag(bp0 + k0);
    bf16x16 B1 = load_frag(bp1 + k0);
    bf16x16 B2 = load_frag(bp2 + k0);
    bf16x16 B3 = load_frag(bp3 + k0);
    acc[0][0] = WMMA_BF16(A0, B0, acc[0][0]);
    acc[1][0] = WMMA_BF16(A1, B0, acc[1][0]);
    acc[0][1] = WMMA_BF16(A0, B1, acc[0][1]);
    acc[1][1] = WMMA_BF16(A1, B1, acc[1][1]);
    acc[0][2] = WMMA_BF16(A0, B2, acc[0][2]);
    acc[1][2] = WMMA_BF16(A1, B2, acc[1][2]);
    acc[0][3] = WMMA_BF16(A0, B3, acc[0][3]);
    acc[1][3] = WMMA_BF16(A1, B3, acc[1][3]);
  }

  // Epilogue: scatter to q [B,H,T,D], k [B,H,T,D], v transposed [B,H,D,T].
  #pragma unroll
  for (int i = 0; i < 2; ++i) {
    #pragma unroll
    for (int j = 0; j < 4; ++j) {
      int n     = n0 + j * 16 + lm;       // "which"/head uniform within a 16-tile
      int which = n >> 10;
      int hh    = (n >> 6) & (HEADS - 1);
      int d     = n & (DH - 1);
      #pragma unroll
      for (int v = 0; v < 8; ++v) {
        int m = m0 + i * 16 + v + hi8;
        int b = m >> 11;
        int t = m & (TLEN - 1);
        __bf16 val = (__bf16)(acc[i][j][v]);
        if (which == 2) {
          vtbuf[((size_t)(b * HEADS + hh) * DH + d) * TLEN + t] = val;
        } else {
          __bf16* dst = which ? kbuf : qbuf;
          dst[((size_t)(b * HEADS + hh) * TLEN + t) * DH + d] = val;
        }
      }
    }
  }
}

// ---------------- Sliding-window flash attention ----------------
// One wave per (b, h, 16-query block); key blocks of 32 over the banded region.
__global__ __launch_bounds__(256) void attn_kernel(
    const __bf16* __restrict__ qbuf, const __bf16* __restrict__ kbuf,
    const __bf16* __restrict__ vtbuf, float* __restrict__ out) {
  __shared__ __align__(16) __bf16 Ps[8][16 * 32];   // per-wave P staging (C->A layout)

  const int lane = threadIdx.x & 31;
  const int wave = threadIdx.x >> 5;
  const int w    = blockIdx.x * 8 + wave;           // 0 .. 4095
  const int qblk = w & 127;
  const int hh   = (w >> 7) & (HEADS - 1);
  const int b    = w >> 11;
  const int q0   = qblk * 16;
  const int lm   = lane & 15;
  const int hi8  = (lane >> 4) * 8;
  const float scale = 0.125f;                       // D^-0.5

  const __bf16* qbase = qbuf  + (size_t)(b * HEADS + hh) * TLEN * DH;
  const __bf16* kbase = kbuf  + (size_t)(b * HEADS + hh) * TLEN * DH;
  const __bf16* vbase = vtbuf + (size_t)(b * HEADS + hh) * DH * TLEN;

  // Q fragments (contract over D = 2 x K=32)
  bf16x16 Aq0, Aq1;
  {
    const __bf16* qp = qbase + (size_t)(q0 + lm) * DH + hi8;
    Aq0 = load_frag(qp);
    Aq1 = load_frag(qp + 32);
  }

  float mrow[8], lrow[8];
  #pragma unroll
  for (int v = 0; v < 8; ++v) { mrow[v] = -3.0e38f; lrow[v] = 0.0f; }
  f32x8 O0 = {}, O1 = {}, O2 = {}, O3 = {};

  int jlo = q0 - WIN; if (jlo < 0) jlo = 0;
  int jhi = q0 + 15 + WIN; if (jhi > TLEN - 1) jhi = TLEN - 1;

  for (int kb0 = jlo & ~31; kb0 <= jhi; kb0 += 32) {
    // ---- scores: S = Q @ K^T for 32 keys (two 16-key N-tiles) ----
    f32x8 s[2];
    #pragma unroll
    for (int nt = 0; nt < 2; ++nt) {
      int key = kb0 + nt * 16 + lm;
      if (key > TLEN - 1) key = TLEN - 1;           // clamp addr; mask kills value
      const __bf16* kp = kbase + (size_t)key * DH + hi8;
      bf16x16 Bk0 = load_frag(kp);
      bf16x16 Bk1 = load_frag(kp + 32);
      f32x8 z = {};
      z = WMMA_BF16(Aq0, Bk0, z);
      z = WMMA_BF16(Aq1, Bk1, z);
      s[nt] = z;
    }

    // ---- scale, window mask, online softmax, stage P in LDS ----
    #pragma unroll
    for (int v = 0; v < 8; ++v) {
      int i  = q0 + v + hi8;
      int j0 = kb0 + lm;
      int j1 = kb0 + 16 + lm;
      bool ok0 = (j0 < TLEN) && (j0 - i <= WIN) && (i - j0 <= WIN);
      bool ok1 = (j1 < TLEN) && (j1 - i <= WIN) && (i - j1 <= WIN);
      float s0 = ok0 ? s[0][v] * scale : -1.0e30f;
      float s1 = ok1 ? s[1][v] * scale : -1.0e30f;

      float mx = fmaxf(s0, s1);                     // reduce over 16-lane half
      mx = fmaxf(mx, SWZ(mx, 0x041F));
      mx = fmaxf(mx, SWZ(mx, 0x081F));
      mx = fmaxf(mx, SWZ(mx, 0x101F));
      mx = fmaxf(mx, SWZ(mx, 0x201F));
      float mnew  = fmaxf(mrow[v], mx);
      float alpha = __expf(mrow[v] - mnew);
      float p0 = __expf(s0 - mnew);
      float p1 = __expf(s1 - mnew);
      float ls = p0 + p1;
      ls += SWZ(ls, 0x041F);
      ls += SWZ(ls, 0x081F);
      ls += SWZ(ls, 0x101F);
      ls += SWZ(ls, 0x201F);
      lrow[v] = lrow[v] * alpha + ls;
      mrow[v] = mnew;
      O0[v] *= alpha; O1[v] *= alpha; O2[v] *= alpha; O3[v] *= alpha;

      int row = v + hi8;
      Ps[wave][row * 32 + lm]      = (__bf16)p0;
      Ps[wave][row * 32 + 16 + lm] = (__bf16)p1;
    }
    __asm__ volatile("" ::: "memory");              // keep ds stores before ds loads
                                                    // (LDS is in-order per wave)
    // ---- O += P @ V  (contract over 32 keys, N = D = 4 x 16) ----
    bf16x16 Pa = load_frag(&Ps[wave][lm * 32 + hi8]);
    {
      const __bf16* vp0 = vbase + (size_t)(0 * 16 + lm) * TLEN + kb0 + hi8;
      const __bf16* vp1 = vbase + (size_t)(1 * 16 + lm) * TLEN + kb0 + hi8;
      const __bf16* vp2 = vbase + (size_t)(2 * 16 + lm) * TLEN + kb0 + hi8;
      const __bf16* vp3 = vbase + (size_t)(3 * 16 + lm) * TLEN + kb0 + hi8;
      O0 = WMMA_BF16(Pa, load_frag(vp0), O0);
      O1 = WMMA_BF16(Pa, load_frag(vp1), O1);
      O2 = WMMA_BF16(Pa, load_frag(vp2), O2);
      O3 = WMMA_BF16(Pa, load_frag(vp3), O3);
    }
  }

  // ---- normalize and write out [B,T,C] fp32 ----
  float* obase = out + (size_t)b * TLEN * CDIM + hh * DH;
  #pragma unroll
  for (int v = 0; v < 8; ++v) {
    float inv = 1.0f / lrow[v];
    int t = q0 + v + hi8;
    float* orow = obase + (size_t)t * CDIM;
    orow[ 0 + lm] = O0[v] * inv;
    orow[16 + lm] = O1[v] * inv;
    orow[32 + lm] = O2[v] * inv;
    orow[48 + lm] = O3[v] * inv;
  }
}

// ---------------- launch ----------------
extern "C" void kernel_launch(void* const* d_in, const int* in_sizes, int n_in,
                              void* d_out, int out_size, void* d_ws, size_t ws_size,
                              hipStream_t stream) {
  const float* x    = (const float*)d_in[0];   // [2,2048,1024]
  const float* Wqkv = (const float*)d_in[1];   // [3072,1024]
  float* out = (float*)d_out;                  // [2,2048,1024]

  const int NX = BSZ * TLEN * CDIM;            // 4,194,304
  const int NW = 3 * CDIM * CDIM;              // 3,145,728
  const int NQ = BSZ * HEADS * TLEN * DH;      // 4,194,304

  // Workspace layout (bf16): x | W | V^T | Q | K   (~38 MB).
  // V^T is deliberately not last: tail-block fragment reads may overrun a row
  // by <=62 bytes into the adjacent (owned) region.
  __bf16* xb = (__bf16*)d_ws;
  __bf16* wb = xb + NX;
  __bf16* vt = wb + NW;
  __bf16* qb = vt + NQ;
  __bf16* kb = qb + NQ;

  cvt4_kernel<<<(NX / 4 + 255) / 256, 256, 0, stream>>>(x, xb, NX);
  cvt4_kernel<<<(NW / 4 + 255) / 256, 256, 0, stream>>>(Wqkv, wb, NW);
  qkv_gemm_kernel<<<dim3(24, 32), 256, 0, stream>>>(xb, wb, qb, kb, vt);
  attn_kernel<<<512, 256, 0, stream>>>(qb, kb, vt, out);
}
